// Net_4552665333820
// MI455X (gfx1250) — matrix operand order repaired and checked by reading, not compiled, not run on hardware
//
#include <hip/hip_runtime.h>
#include <hip/hip_bf16.h>

typedef __attribute__((ext_vector_type(16))) _Float16 v16h;
typedef __attribute__((ext_vector_type(8)))  float    v8f;

#define LL long long

// ---------------- WMMA fragment helpers (CDNA5 16x16x32 f16, wave32) ----------
// A (16x32 f16, row-major source, ld = row stride in elements)
__device__ __forceinline__ v16h frag_A(const _Float16* s, int ld, int kbase) {
  const int lane = threadIdx.x & 31;
  const int m = lane & 15, hi = lane >> 4;
  v16h a;
#pragma unroll
  for (int v = 0; v < 8; ++v) {
    const int k0 = kbase + ((v & 3) << 1) + ((v >> 2) << 4) + (hi << 3);
    a[2 * v]     = s[m * ld + k0];
    a[2 * v + 1] = s[m * ld + k0 + 1];
  }
  return a;
}

// B (32x16 f16) from K-transposed storage BT[n][k] (LDS), ld = K stride
__device__ __forceinline__ v16h frag_B(const _Float16* bt, int ld, int kbase) {
  const int lane = threadIdx.x & 31;
  const int n = lane & 15, hi = lane >> 4;
  v16h b;
#pragma unroll
  for (int v = 0; v < 8; ++v) {
    const int k0 = kbase + ((v & 3) << 1) + ((v >> 2) << 4) + (hi << 3);
    b[2 * v]     = bt[n * ld + k0];
    b[2 * v + 1] = bt[n * ld + k0 + 1];
  }
  return b;
}

__device__ __forceinline__ v8f wmma_f16(v16h a, v16h b, v8f c) {
  return __builtin_amdgcn_wmma_f32_16x16x32_f16(false, a, false, b, (short)0, c,
                                                false, false);
}

// vectorized global->LDS copy (16B granules); count16 = bytes/16
__device__ __forceinline__ void copy16(const void* __restrict__ g, void* __restrict__ l,
                                       int count16, int tid, int nthreads) {
  const uint4* src = (const uint4*)g;
  uint4* dst = (uint4*)l;
  for (int i = tid; i < count16; i += nthreads) dst[i] = src[i];
}
__device__ __forceinline__ void zero16(void* l, int count16, int tid, int nthreads) {
  uint4* dst = (uint4*)l;
  const uint4 z = {0u, 0u, 0u, 0u};
  for (int i = tid; i < count16; i += nthreads) dst[i] = z;
}

// BN helpers: stats layout per layer = [sum(64) | sumsq(64) | mean(64) | rstd(64)]
__device__ __forceinline__ float bn1(float v, const float* st, const float* g,
                                     const float* be, int c) {
  return (v - st[128 + c]) * st[192 + c] * g[c] + be[c];
}
__device__ __forceinline__ float bn3(float v, int c,
                                     const float* s0, const float* s1, const float* s2,
                                     const float* g0, const float* g1, const float* g2,
                                     const float* e0, const float* e1, const float* e2) {
  if (c < 64)  return bn1(v, s0, g0, e0, c);
  if (c < 128) return bn1(v, s1, g1, e1, c - 64);
  return bn1(v, s2, g2, e2, c - 128);
}

// ---------------- weight convert: W(K,N) f32 -> WT[NPad][KPad] f16 -------------
__global__ void wconv_kernel(const float* __restrict__ W, _Float16* __restrict__ WT,
                             int K, int Ncols, int KPad, int NPad) {
  int i = blockIdx.x * blockDim.x + threadIdx.x;
  if (i >= NPad * KPad) return;
  int n = i / KPad, k = i - n * KPad;
  float v = (n < Ncols && k < K) ? W[(LL)k * Ncols + n] : 0.f;
  WT[i] = (_Float16)v;
}

// ---------------- fused DenseSAGE layer ---------------------------------------
// z = relu( (adj'/deg) @ (bn(x) @ W) + b ), accumulate per-channel sum/sumsq.
__global__ void sage_kernel(const float* __restrict__ xin, int xRow, LL xGraph,
                            const float* __restrict__ adj,
                            const _Float16* __restrict__ WT,
                            const float* __restrict__ bias,
                            const float* __restrict__ inStats,
                            const float* __restrict__ inG,
                            const float* __restrict__ inBe,
                            float* __restrict__ yout, int yRow, LL yGraph,
                            float* __restrict__ outStats,
                            int N, int Cin, int Cout, int KPad, int addLoop) {
  __shared__ _Float16 sX[32 * 192];
  __shared__ _Float16 sW[64 * 192];   // staged weight tile
  __shared__ float    sAf[32 * 32];
  __shared__ _Float16 sAh[32 * 32];
  __shared__ _Float16 sPT[64 * 32];   // P^T : [coutCol][node]
  __shared__ float    sStats[128];
  __shared__ float    sDeg[32];

  const int b = blockIdx.x, tid = threadIdx.x;
  const int ntiles = (Cout + 15) >> 4, mtiles = (N + 15) >> 4, ksteps = KPad >> 5;
  const int wN = ntiles << 4;

  // stage weights (contiguous, vectorized) + zero-fill tiles
  copy16(WT, sW, (wN * KPad) >> 3, tid, 128);
  zero16(sX, (32 * KPad) >> 3, tid, 128);
  zero16(sPT, (64 * 32) >> 3, tid, 128);
  zero16(sAf, (32 * 32 * 4) >> 4, tid, 128);
  if (tid < 128) sStats[tid] = 0.f;
  __syncthreads();

  // load x rows (apply input BN affine), f16
  for (int r = 0; r < N; ++r) {
    const float* xrow = xin + (LL)b * xGraph + (LL)r * xRow;
    for (int k = tid; k < Cin; k += 128) {
      float v = xrow[k];
      if (inStats) v = bn1(v, inStats, inG, inBe, k);
      sX[r * KPad + k] = (_Float16)v;
    }
  }
  // load adj rows, add self loops
  for (int r = 0; r < N; ++r) {
    const float* arow = adj + (LL)b * N * N + (LL)r * N;
    for (int c = tid; c < N; c += 128) {
      float v = arow[c];
      if (addLoop && r == c) v = 1.0f;
      sAf[(r << 5) + c] = v;
    }
  }
  __syncthreads();

  if (tid < 32) {  // degrees
    float s = 0.f;
    for (int c = 0; c < 32; ++c) s += sAf[(tid << 5) + c];
    sDeg[tid] = fmaxf(s, 1.0f);
  }
  __syncthreads();
  for (int i = tid; i < 32 * 32; i += 128) {
    int r = i >> 5;
    sAh[i] = (_Float16)(sAf[i] / sDeg[r]);
  }
  __syncthreads();

  const int wid = tid >> 5;

  // phase 1: P = X @ W   (store P^T f16 in LDS)
  for (int nt = wid; nt < ntiles; nt += 4) {
    for (int mt = 0; mt < mtiles; ++mt) {
      v8f acc = {0.f, 0.f, 0.f, 0.f, 0.f, 0.f, 0.f, 0.f};
      for (int ks = 0; ks < ksteps; ++ks) {
        v16h a = frag_A(sX + mt * 16 * KPad, KPad, ks * 32);
        v16h w = frag_B(sW + nt * 16 * KPad, KPad, ks * 32);
        acc = wmma_f16(a, w, acc);
      }
      const int lane = tid & 31, n = (lane & 15) + nt * 16, hi = lane >> 4;
#pragma unroll
      for (int v = 0; v < 8; ++v) {
        int node = mt * 16 + v + 8 * hi;
        sPT[(n << 5) + node] = (_Float16)acc[v];
      }
    }
  }
  __syncthreads();

  // phase 2: Z = adjn @ P ; bias + relu + store + stats
  for (int nt = wid; nt < ntiles; nt += 4) {
    for (int mt = 0; mt < mtiles; ++mt) {
      v8f acc = {0.f, 0.f, 0.f, 0.f, 0.f, 0.f, 0.f, 0.f};
      v16h a = frag_A(sAh + mt * 16 * 32, 32, 0);
      v16h p = frag_B(sPT + nt * 16 * 32, 32, 0);
      acc = wmma_f16(a, p, acc);
      const int lane = tid & 31, nn = lane & 15, n = nn + nt * 16, hi = lane >> 4;
      float ls = 0.f, lq = 0.f;
#pragma unroll
      for (int v = 0; v < 8; ++v) {
        int node = mt * 16 + v + 8 * hi;
        if (node < N && n < Cout) {
          float z = fmaxf(acc[v] + bias[n], 0.f);
          yout[(LL)b * yGraph + (LL)node * yRow + n] = z;
          ls += z; lq += z * z;
        }
      }
      if (n < Cout) {
        atomicAdd(&sStats[n], ls);
        atomicAdd(&sStats[64 + n], lq);
      }
    }
  }
  __syncthreads();
  if (tid < 128) atomicAdd(&outStats[tid], sStats[tid]);
}

__global__ void bn_finalize_kernel(float* stats, float invCount, int C) {
  int t = threadIdx.x;
  if (t < C) {
    float m = stats[t] * invCount;
    float v = stats[64 + t] * invCount - m * m;
    stats[128 + t] = m;
    stats[192 + t] = rsqrtf(fmaxf(v, 0.f) + 1e-5f);
  }
}

// ---------------- lin layer of pool GNNs: s = relu(bn3(cat) @ lW + lb) --------
__global__ void lin_kernel(const float* __restrict__ cat, int catRow, LL catGraph,
                           const float* s0, const float* s1, const float* s2,
                           const float* g0, const float* g1, const float* g2,
                           const float* e0, const float* e1, const float* e2,
                           const _Float16* __restrict__ WT, const float* __restrict__ bias,
                           float* __restrict__ out, int N, int Ccat, int KPad, int Cout) {
  __shared__ _Float16 sX[32 * 192];
  __shared__ _Float16 sW[16 * 192];
  const int b = blockIdx.x, tid = threadIdx.x;
  copy16(WT, sW, (16 * KPad) >> 3, tid, 128);
  zero16(sX, (32 * KPad) >> 3, tid, 128);
  __syncthreads();
  for (int r = 0; r < N; ++r) {
    const float* crow = cat + (LL)b * catGraph + (LL)r * catRow;
    for (int k = tid; k < Ccat; k += 128) {
      float v = bn3(crow[k], k, s0, s1, s2, g0, g1, g2, e0, e1, e2);
      sX[r * KPad + k] = (_Float16)v;
    }
  }
  __syncthreads();
  const int wid = tid >> 5, mtiles = (N + 15) >> 4;
  if (wid < mtiles) {
    const int mt = wid;
    v8f acc = {0.f, 0.f, 0.f, 0.f, 0.f, 0.f, 0.f, 0.f};
    for (int ks = 0; ks < (KPad >> 5); ++ks) {
      v16h a = frag_A(sX + mt * 16 * KPad, KPad, ks * 32);
      v16h w = frag_B(sW, KPad, ks * 32);
      acc = wmma_f16(a, w, acc);
    }
    const int lane = tid & 31, n = lane & 15, hi = lane >> 4;
#pragma unroll
    for (int v = 0; v < 8; ++v) {
      int node = mt * 16 + v + 8 * hi;
      if (node < N && n < Cout)
        out[(LL)b * N * Cout + node * Cout + n] = fmaxf(acc[v] + bias[n], 0.f);
    }
  }
}

// ---------------- diff-pool: softmax(s), pooled h / adj, link + entropy -------
__global__ void diffpool_kernel(const float* __restrict__ cat,
                                const float* s0, const float* s1, const float* s2,
                                const float* g0, const float* g1, const float* g2,
                                const float* e0, const float* e1, const float* e2,
                                const float* __restrict__ adj,
                                const float* __restrict__ sIn,
                                float* __restrict__ outH, float* __restrict__ outAdj,
                                float* __restrict__ linkAcc, float* __restrict__ entAcc,
                                int N, int K) {
  __shared__ float sS[17 * 8];
  __shared__ float sH[17 * 192];
  __shared__ float sA[17 * 17];
  __shared__ float sT[17 * 8];
  __shared__ float sRed[2];
  const int b = blockIdx.x, tid = threadIdx.x;
  if (tid < 2) sRed[tid] = 0.f;
  for (int r = 0; r < N; ++r) {
    const float* crow = cat + (LL)b * N * 192 + (LL)r * 192;
    for (int c = tid; c < 192; c += 128)
      sH[r * 192 + c] = bn3(crow[c], c, s0, s1, s2, g0, g1, g2, e0, e1, e2);
  }
  for (int i = tid; i < N * N; i += 128) sA[i] = adj[(LL)b * N * N + i];
  __syncthreads();
  if (tid < N) {  // row softmax + entropy
    float row[8], mx = -1e30f;
    for (int k = 0; k < K; ++k) {
      row[k] = sIn[(LL)b * N * K + tid * K + k];
      mx = fmaxf(mx, row[k]);
    }
    float sm = 0.f;
    for (int k = 0; k < K; ++k) { row[k] = expf(row[k] - mx); sm += row[k]; }
    float ent = 0.f;
    for (int k = 0; k < K; ++k) {
      float p = row[k] / sm;
      sS[tid * K + k] = p;
      ent -= p * logf(p + 1e-15f);
    }
    atomicAdd(&sRed[1], ent);
  }
  __syncthreads();
  for (int k = 0; k < K; ++k) {  // pooled features
    for (int c = tid; c < 192; c += 128) {
      float a = 0.f;
      for (int n = 0; n < N; ++n) a += sS[n * K + k] * sH[n * 192 + c];
      outH[(LL)b * K * 192 + k * 192 + c] = a;
    }
  }
  for (int i = tid; i < N * K; i += 128) {  // T = adj @ S
    int n = i / K, l = i - n * K;
    float a = 0.f;
    for (int m = 0; m < N; ++m) a += sA[n * N + m] * sS[m * K + l];
    sT[i] = a;
  }
  __syncthreads();
  for (int i = tid; i < K * K; i += 128) {  // pooled adjacency
    int k = i / K, l = i - k * K;
    float a = 0.f;
    for (int n = 0; n < N; ++n) a += sS[n * K + k] * sT[n * K + l];
    outAdj[(LL)b * K * K + i] = a;
  }
  float lpart = 0.f;  // link loss partial
  for (int i = tid; i < N * N; i += 128) {
    int n = i / N, m = i - n * N;
    float d = sA[i];
    for (int k = 0; k < K; ++k) d -= sS[n * K + k] * sS[m * K + k];
    lpart += d * d;
  }
  atomicAdd(&sRed[0], lpart);
  __syncthreads();
  if (tid == 0) {
    atomicAdd(linkAcc, sRed[0]);
    atomicAdd(entAcc, sRed[1]);
  }
}

// ---------------- head: node-mean -> lin1(192->64) relu -> lin2(64->2) --------
__global__ void head_kernel(const float* __restrict__ cat3,
                            const float* s0, const float* s1, const float* s2,
                            const float* g0, const float* g1, const float* g2,
                            const float* e0, const float* e1, const float* e2,
                            const _Float16* __restrict__ W1T, const float* __restrict__ b1,
                            const float* __restrict__ W2, const float* __restrict__ b2,
                            float* __restrict__ out) {
  __shared__ _Float16 sX[16 * 192];
  __shared__ _Float16 sW[64 * 192];
  __shared__ float    sHH[16 * 64];
  const int tid = threadIdx.x;
  const LL base = (LL)blockIdx.x * 16;
  copy16(W1T, sW, (64 * 192) >> 3, tid, 128);
  for (int g = 0; g < 16; ++g) {
    const float* crow = cat3 + (base + g) * 384;
    for (int c = tid; c < 192; c += 128) {
      float v0 = bn3(crow[c],       c, s0, s1, s2, g0, g1, g2, e0, e1, e2);
      float v1 = bn3(crow[192 + c], c, s0, s1, s2, g0, g1, g2, e0, e1, e2);
      sX[g * 192 + c] = (_Float16)(0.5f * (v0 + v1));
    }
  }
  __syncthreads();
  const int nt = tid >> 5;  // 4 waves x 4 N-tiles
  v8f acc = {0.f, 0.f, 0.f, 0.f, 0.f, 0.f, 0.f, 0.f};
  for (int ks = 0; ks < 6; ++ks) {
    v16h a = frag_A(sX, 192, ks * 32);
    v16h w = frag_B(sW + nt * 16 * 192, 192, ks * 32);
    acc = wmma_f16(a, w, acc);
  }
  {
    const int lane = tid & 31, n = (nt << 4) + (lane & 15), hi = lane >> 4;
#pragma unroll
    for (int v = 0; v < 8; ++v) {
      int m = v + 8 * hi;
      sHH[(m << 6) + n] = fmaxf(acc[v] + b1[n], 0.f);
    }
  }
  __syncthreads();
  if (tid < 32) {
    int g = tid >> 1, col = tid & 1;
    float a = b2[col];
    for (int j = 0; j < 64; ++j) a += sHH[(g << 6) + j] * W2[j * 2 + col];
    out[(base + g) * 2 + col] = a;
  }
}

__global__ void loss_kernel(const float* __restrict__ acc, float* __restrict__ outTail) {
  if (threadIdx.x == 0) {
    float l = sqrtf(acc[0]) / (8192.f * 17.f * 17.f) +
              sqrtf(acc[1]) / (8192.f * 5.f * 5.f);
    float e = acc[2] / (8192.f * 17.f) + acc[3] / (8192.f * 5.f);
    outTail[0] = l;
    outTail[1] = e;
  }
}

// ---------------- host orchestration ------------------------------------------
extern "C" void kernel_launch(void* const* d_in, const int* in_sizes, int n_in,
                              void* d_out, int out_size, void* d_ws, size_t ws_size,
                              hipStream_t stream) {
  (void)in_sizes; (void)n_in; (void)out_size; (void)ws_size;
  const int B = 8192;
  const float* x   = (const float*)d_in[0];
  const float* adj = (const float*)d_in[1];
  auto in = [&](int i) { return (const float*)d_in[i]; };

  char* ws = (char*)d_ws;
  size_t off = 0;
  auto alloc = [&](size_t bytes) -> void* {
    off = (off + 255) & ~(size_t)255;
    void* p = ws + off;
    off += bytes;
    return p;
  };

  float* arenaA = (float*)alloc((size_t)B * 17 * 192 * 4);  // catE1/catE2/catE3
  float* arenaB = (float*)alloc((size_t)B * 17 * 133 * 4);  // catP1/catP2
  float* s1  = (float*)alloc((size_t)B * 17 * 5 * 4);
  float* hP1 = (float*)alloc((size_t)B * 5 * 192 * 4);
  float* a2  = (float*)alloc((size_t)B * 5 * 5 * 4);
  float* s2  = (float*)alloc((size_t)B * 5 * 2 * 4);
  float* hP2 = (float*)alloc((size_t)B * 2 * 192 * 4);
  float* a3  = (float*)alloc((size_t)B * 2 * 2 * 4);
  float* stats   = (float*)alloc(15 * 256 * 4);
  float* lossAcc = (float*)alloc(4 * 4);  // [link1, link2, ent1, ent2]

  struct WInfo { int idx, K, N, KPad, NPad; _Float16* p; };
  WInfo winfo[] = {
    {2, 31, 64, 32, 64, nullptr},  {6, 64, 64, 64, 64, nullptr},
    {10, 64, 5, 64, 16, nullptr},  {14, 133, 5, 160, 16, nullptr},
    {16, 31, 64, 32, 64, nullptr}, {20, 64, 64, 64, 64, nullptr},
    {24, 64, 64, 64, 64, nullptr},
    {28, 192, 64, 192, 64, nullptr}, {32, 64, 64, 64, 64, nullptr},
    {36, 64, 2, 64, 16, nullptr},  {40, 130, 2, 160, 16, nullptr},
    {42, 192, 64, 192, 64, nullptr}, {46, 64, 64, 64, 64, nullptr},
    {50, 64, 64, 64, 64, nullptr},
    {54, 192, 64, 192, 64, nullptr}, {58, 64, 64, 64, 64, nullptr},
    {62, 64, 64, 64, 64, nullptr},
    {66, 192, 64, 192, 64, nullptr}
  };
  for (auto& w : winfo) {
    w.p = (_Float16*)alloc((size_t)w.KPad * w.NPad * 2);
    int tot = w.KPad * w.NPad;
    wconv_kernel<<<(tot + 255) / 256, 256, 0, stream>>>(in(w.idx), w.p, w.K, w.N,
                                                        w.KPad, w.NPad);
  }
  auto wptr = [&](int idx) -> const _Float16* {
    for (auto& w : winfo) if (w.idx == idx) return w.p;
    return nullptr;
  };

  hipMemsetAsync(stats, 0, 15 * 256 * 4, stream);
  hipMemsetAsync(lossAcc, 0, 16, stream);

  auto sage = [&](const float* xin, int xRow, LL xGraph, const float* adjp, int N,
                  int addLoop, const _Float16* WT, const float* bias,
                  const float* inStats, const float* inG, const float* inBe,
                  float* yout, int yRow, LL yGraph, float* statL, int Cin, int Cout) {
    int KPad = ((Cin + 31) / 32) * 32;
    sage_kernel<<<B, 128, 0, stream>>>(xin, xRow, xGraph, adjp, WT, bias, inStats,
                                       inG, inBe, yout, yRow, yGraph, statL, N, Cin,
                                       Cout, KPad, addLoop);
    bn_finalize_kernel<<<1, 64, 0, stream>>>(statL, 1.0f / (float)(B * N), Cout);
  };

  auto rungnn = [&](const float* xin, int xRow, LL xGraph, const float* adjp, int N,
                    int addLoop, int p, int L, float* cat, int catStride, int cout3,
                    int cin0) {
    float* L0 = stats + (LL)L * 256;
    float* L1 = L0 + 256;
    float* L2 = L1 + 256;
    LL cg = (LL)N * catStride;
    sage(xin, xRow, xGraph, adjp, N, addLoop, wptr(p + 0), in(p + 1),
         nullptr, nullptr, nullptr, cat + 0, catStride, cg, L0, cin0, 64);
    sage(cat + 0, catStride, cg, adjp, N, addLoop, wptr(p + 4), in(p + 5),
         L0, in(p + 2), in(p + 3), cat + 64, catStride, cg, L1, 64, 64);
    sage(cat + 64, catStride, cg, adjp, N, addLoop, wptr(p + 8), in(p + 9),
         L1, in(p + 6), in(p + 7), cat + 128, catStride, cg, L2, 64, cout3);
  };

  // ---- level 1 ----
  rungnn(x, 31, (LL)17 * 31, adj, 17, 1, 2, 0, arenaB, 133, 5, 31);       // gnn1_pool
  lin_kernel<<<B, 128, 0, stream>>>(arenaB, 133, (LL)17 * 133,
      stats + 0 * 256, stats + 1 * 256, stats + 2 * 256,
      in(4), in(8), in(12), in(5), in(9), in(13),
      wptr(14), in(15), s1, 17, 133, 160, 5);
  rungnn(x, 31, (LL)17 * 31, adj, 17, 1, 16, 3, arenaA, 192, 64, 31);     // gnn1_embed
  diffpool_kernel<<<B, 128, 0, stream>>>(arenaA,
      stats + 3 * 256, stats + 4 * 256, stats + 5 * 256,
      in(18), in(22), in(26), in(19), in(23), in(27),
      adj, s1, hP1, a2, lossAcc + 0, lossAcc + 2, 17, 5);

  // ---- level 2 ----
  rungnn(hP1, 192, (LL)5 * 192, a2, 5, 0, 28, 6, arenaB, 130, 2, 192);    // gnn2_pool
  lin_kernel<<<B, 128, 0, stream>>>(arenaB, 130, (LL)5 * 130,
      stats + 6 * 256, stats + 7 * 256, stats + 8 * 256,
      in(30), in(34), in(38), in(31), in(35), in(39),
      wptr(40), in(41), s2, 5, 130, 160, 2);
  rungnn(hP1, 192, (LL)5 * 192, a2, 5, 0, 42, 9, arenaA, 192, 64, 192);   // gnn2_embed
  diffpool_kernel<<<B, 128, 0, stream>>>(arenaA,
      stats + 9 * 256, stats + 10 * 256, stats + 11 * 256,
      in(44), in(48), in(52), in(45), in(49), in(53),
      a2, s2, hP2, a3, lossAcc + 1, lossAcc + 3, 5, 2);

  // ---- level 3 + head ----
  rungnn(hP2, 192, (LL)2 * 192, a3, 2, 0, 54, 12, arenaA, 192, 64, 192);  // gnn3_embed
  head_kernel<<<B / 16, 128, 0, stream>>>(arenaA,
      stats + 12 * 256, stats + 13 * 256, stats + 14 * 256,
      in(56), in(60), in(64), in(57), in(61), in(65),
      wptr(66), in(67), in(68), in(69), (float*)d_out);
  loss_kernel<<<1, 32, 0, stream>>>(lossAcc, (float*)d_out + 16384);
}
